// CNN_88098369175808
// MI455X (gfx1250) — matrix operand order repaired and checked by reading, not compiled, not run on hardware
//
#include <hip/hip_runtime.h>
#include <hip/hip_bf16.h>

typedef __attribute__((ext_vector_type(16))) _Float16 v16h;
typedef __attribute__((ext_vector_type(8)))  float    v8f;
typedef __attribute__((ext_vector_type(4)))  int      v4i;

#define T_LEN    8192
#define N_BATCH  64
#define CIN      64
#define HID      10
#define HPAD     16
#define OUTC     2
#define CHUNK    64                 // timesteps staged per LDS chunk
#define NCHUNK   (T_LEN / CHUNK)

#define F_VTH    0.2f
#define F_DECAY  0.25f
#define K_MEM    0.1f               // DT * TAU_MEM_INV
#define K_SYN    0.2f               // DT * TAU_SYN_INV
#define K_ADAPT  (0.001f / 700.0f)  // DT * TAU_ADAPT_INV

__device__ __forceinline__ unsigned ballot32(bool p) {
#if __has_builtin(__builtin_amdgcn_ballot_w32)
  return __builtin_amdgcn_ballot_w32(p);
#else
  return (unsigned)__ballot(p);
#endif
}

#if __has_builtin(__builtin_amdgcn_global_load_async_to_lds_b128)
#define HAVE_ASYNC_LDS 1
typedef __attribute__((address_space(1))) v4i gas_v4i;
typedef __attribute__((address_space(3))) v4i las_v4i;
__device__ __forceinline__ void async_cp16(const float* g, float* l) {
  __builtin_amdgcn_global_load_async_to_lds_b128(
      (gas_v4i*)g, (las_v4i*)l, 0, 0);
}
__device__ __forceinline__ void async_wait0() {
#if __has_builtin(__builtin_amdgcn_s_wait_asynccnt)
  __builtin_amdgcn_s_wait_asynccnt(0);
#else
  asm volatile("s_wait_asynccnt 0" ::: "memory");
#endif
  asm volatile("" ::: "memory");   // keep LDS reads below the wait
}
#else
#define HAVE_ASYNC_LDS 0
#endif

// ---------------------------------------------------------------------------
// Kernel 1: front projection GEMM via WMMA.
//   proj[b][t][n] = sum_c w_front[n][c] * x[b][0][c+1][t] + b_front[n]
// One wave per (b, 16-timestep tile): D(16x16) += A(16x32) x B(32x16), K=64
// as two v_wmma_f32_16x16x32_f16. M = t, N = h, K = c.
// Pad rows n=10..15 are CLAMPED duplicates of valid rows (never consumed by
// the scan, which masks all reductions to j<10) -> zero predication, zero
// divergence, fully pipelined loads.
// ---------------------------------------------------------------------------
__global__ __launch_bounds__(256) void front_gemm_wmma(
    const float* __restrict__ x, const float* __restrict__ w_front,
    const float* __restrict__ b_front, float* __restrict__ proj)
{
  const int lane   = threadIdx.x & 31;
  const int wid    = blockIdx.x * 8 + (threadIdx.x >> 5);
  const int b      = wid >> 9;              // 512 tiles per batch
  const int t0     = (wid & 511) * 16;

  const int  n   = lane & 15;               // N index (hidden unit)
  const int  m   = lane & 15;               // M index (timestep within tile)
  const bool hi  = lane >= 16;

  const int nw = (n < HID) ? n : (HID - 1); // clamped row (pad cols unused)
  const float* wrow = w_front + nw * CIN;

  // x slice: channels 1..64 of the 66-channel input
  const float* xp = x + ((size_t)(b * 66 + 1)) * T_LEN + t0;

  v8f acc = {};
#pragma unroll
  for (int c0 = 0; c0 < CIN; c0 += 32) {
    // ---- B fragment: 16 contiguous floats of one w_front row -> 4x b128
    const float4* w4 = (const float4*)(wrow + c0 + (hi ? 16 : 0));
    const float4 w0 = w4[0], w1 = w4[1], w2 = w4[2], w3 = w4[3];
    v16h w;
    w[0]  = (_Float16)w0.x;  w[1]  = (_Float16)w0.y;
    w[2]  = (_Float16)w0.z;  w[3]  = (_Float16)w0.w;
    w[4]  = (_Float16)w1.x;  w[5]  = (_Float16)w1.y;
    w[6]  = (_Float16)w1.z;  w[7]  = (_Float16)w1.w;
    w[8]  = (_Float16)w2.x;  w[9]  = (_Float16)w2.y;
    w[10] = (_Float16)w2.z;  w[11] = (_Float16)w2.w;
    w[12] = (_Float16)w3.x;  w[13] = (_Float16)w3.y;
    w[14] = (_Float16)w3.z;  w[15] = (_Float16)w3.w;

    // ---- A fragment: 16 b32 gathers (channel stride = T_LEN)
    v16h a;
#pragma unroll
    for (int vv = 0; vv < 8; ++vv) {
#pragma unroll
      for (int p = 0; p < 2; ++p) {
        // A element s = 2v+p holds K = s (+16 hi), matching contiguous B
        const int ka = 2 * vv + p + (hi ? 16 : 0);
        a[2 * vv + p] = (_Float16)xp[(size_t)(c0 + ka) * T_LEN + m];
      }
    }

    acc = __builtin_amdgcn_wmma_f32_16x16x32_f16(
        /*neg_a=*/false, a, /*neg_b=*/false, w,
        /*c_mod=*/(short)0, acc, /*reuse_a=*/false, /*reuse_b=*/false);
  }

  const float bias = b_front[nw];
  float* op = proj + ((size_t)b * T_LEN + t0) * HPAD + n;
#pragma unroll
  for (int r = 0; r < 8; ++r) {
    const int mm = r + (hi ? 8 : 0);        // D: lanes 0-15 -> M=r, hi -> M=8+r
    op[(size_t)mm * HPAD] = acc[r] + bias;
  }
}

// ---------------------------------------------------------------------------
// Kernel 2: fused front-LIF + LSNN recurrent scan + classifier LIF + mean.
// One wave per batch; lane h (<10) owns hidden unit h. Spike matvecs are done
// as ballot-mask selected sums (all reductions mask j<10, so pad-lane state is
// never consumed). proj is streamed through a double-buffered LDS chunk using
// async global->LDS loads.
// ---------------------------------------------------------------------------
__global__ __launch_bounds__(32) void lsnn_scan(
    const float* __restrict__ proj,
    const float* __restrict__ w_in,  const float* __restrict__ w_rec,
    const float* __restrict__ w_cls, const float* __restrict__ b_cls,
    float* __restrict__ out)
{
  __shared__ float sbuf[2][CHUNK * HPAD];   // 2 x 4 KB

  const int b    = blockIdx.x;
  const int lane = threadIdx.x & 31;
  const int h    = lane & 15;

  // Per-lane weight rows (zero padded beyond valid lanes)
  float wi[HID], wr[HID], wc[HID];
#pragma unroll
  for (int j = 0; j < HID; ++j) {
    wi[j] = (lane < HID)  ? w_in [lane * HID + j] : 0.0f;
    wr[j] = (lane < HID)  ? w_rec[lane * HID + j] : 0.0f;
    wc[j] = (lane < OUTC) ? w_cls[lane * HID + j] : 0.0f;
  }
  const float bc = (lane < OUTC) ? b_cls[lane] : 0.0f;

  const float* pb = proj + (size_t)b * T_LEN * HPAD;

  auto stage = [&](int buf, int ck) {
    const float* g = pb + (size_t)ck * CHUNK * HPAD;
#pragma unroll
    for (int q = 0; q < (CHUNK * HPAD) / (32 * 4); ++q) {  // 8 x b128 per lane
      const int e4 = lane + 32 * q;
#if HAVE_ASYNC_LDS
      async_cp16(g + e4 * 4, &sbuf[buf][e4 * 4]);
#else
      const float4 v4 = ((const float4*)g)[e4];
      ((float4*)&sbuf[buf][0])[e4] = v4;
#endif
    }
  };

  stage(0, 0);
#if HAVE_ASYNC_LDS
  async_wait0();
#endif

  // State (reference init): front LIF u,o = 0; lsnn z,v,i = 0, b = VTH;
  // classifier LIF u2,o2 = 0.
  float u = 0.0f, o = 0.0f;
  float v = 0.0f, cur = 0.0f, bb = F_VTH;
  float u2 = 0.0f, o2 = 0.0f, acc = 0.0f;
  unsigned mz = 0u;                          // z spike mask (bits 0..9)

  for (int ck = 0; ck < NCHUNK; ++ck) {
    const int curbuf = ck & 1;
    if (ck + 1 < NCHUNK) stage(curbuf ^ 1, ck + 1);   // overlap next chunk load
    const float* sb = &sbuf[curbuf][0];

#pragma unroll 8
    for (int s = 0; s < CHUNK; ++s) {
      // ---- front LIF (pad lanes mimic valid lanes; bits >=10 masked below)
      const float p = sb[s * HPAD + h];
      u = F_DECAY * u * (1.0f - o) + p;
      const bool sp = (u - F_VTH) > 0.0f;
      o = sp ? 1.0f : 0.0f;
      const unsigned ms = ballot32(sp);

      // ---- x_in = s @ w_in.T ; rec = z @ w_rec.T  (mask-selected sums)
      float xin = 0.0f, rec = 0.0f;
#pragma unroll
      for (int j = 0; j < HID; ++j) {
        if (ms & (1u << j)) xin += wi[j];
        if (mz & (1u << j)) rec += wr[j];
      }

      // ---- LSNN neuron update
      const float ij = cur + xin + rec;
      const float vd = v + K_MEM * (ij - v);
      cur = ij - K_SYN * ij;
      bb  = bb + K_ADAPT * (F_VTH - bb);     // stays VTH (BETA=0), kept for fidelity
      const bool zs = (vd - bb) > 0.0f;
      mz = ballot32(zs);
      v  = zs ? 0.0f : vd;

      // ---- classifier: w_cls @ z + b_cls, then LIF + accumulate
      float co = bc;
#pragma unroll
      for (int j = 0; j < HID; ++j)
        if (mz & (1u << j)) co += wc[j];
      u2 = F_DECAY * u2 * (1.0f - o2) + co;
      const bool s2 = (u2 - F_VTH) > 0.0f;
      o2 = s2 ? 1.0f : 0.0f;
      acc += o2;
    }
#if HAVE_ASYNC_LDS
    async_wait0();                            // next chunk resident before swap
#endif
  }

  if (lane < OUTC) out[b * OUTC + lane] = acc * (1.0f / (float)T_LEN);
}

// ---------------------------------------------------------------------------
extern "C" void kernel_launch(void* const* d_in, const int* in_sizes, int n_in,
                              void* d_out, int out_size, void* d_ws, size_t ws_size,
                              hipStream_t stream) {
  const float* x       = (const float*)d_in[0];
  const float* w_front = (const float*)d_in[1];
  const float* b_front = (const float*)d_in[2];
  const float* w_in    = (const float*)d_in[3];
  const float* w_rec   = (const float*)d_in[4];
  const float* w_cls   = (const float*)d_in[5];
  const float* b_cls   = (const float*)d_in[6];

  float* proj = (float*)d_ws;   // N_BATCH * T_LEN * HPAD floats = 33.5 MB

  // 64 batches * 512 tiles / 8 waves per block
  front_gemm_wmma<<<(N_BATCH * (T_LEN / 16)) / 8, 256, 0, stream>>>(
      x, w_front, b_front, proj);

  lsnn_scan<<<N_BATCH, 32, 0, stream>>>(proj, w_in, w_rec, w_cls, b_cls,
                                        (float*)d_out);
}
